// CrossAttention_67705864454178
// MI455X (gfx1250) — matrix operand order repaired and checked by reading, not compiled, not run on hardware
//
#include <hip/hip_runtime.h>
#include <hip/hip_bf16.h>

typedef __attribute__((ext_vector_type(16))) _Float16 v16h;
typedef __attribute__((ext_vector_type(8)))  _Float16 v8h;
typedef __attribute__((ext_vector_type(8)))  float    v8f;

#define NUM_HEADS 12
#define BDIM 768
#define HD 64
#define NB 8
#define N1 64
#define N2 4096
#define SCALE 0.125f

// LDS B-panel slab: 64 N-rows x 256 K-halves, padded row stride
#define KC 256
#define BPAD 8
#define BSTRIDE (KC + BPAD)   // 264 halves = 132 dwords; 132 % 64 = 4 -> conflict-free

__device__ __forceinline__ v8f wmma_f16(v16h a, v16h b, v8f c) {
    return __builtin_amdgcn_wmma_f32_16x16x32_f16(
        /*neg_a=*/false, a, /*neg_b=*/false, b,
        /*c_mod=*/(short)0, c, /*reuse_a=*/false, /*reuse_b=*/false);
}

__device__ __forceinline__ v16h cat8(v8h lo, v8h hi) {
    return __builtin_shufflevector(lo, hi, 0,1,2,3,4,5,6,7,8,9,10,11,12,13,14,15);
}

// Async global->LDS 16-byte copy (ASYNCcnt-tracked, no VGPR round trip)
__device__ __forceinline__ void async_copy_b128(const void* gptr, void* lptr) {
    uint64_t g = (uint64_t)(uintptr_t)gptr;
    uint32_t l = (uint32_t)(uintptr_t)lptr;   // LDS aperture: low 32 bits = LDS byte offset
    asm volatile("global_load_async_to_lds_b128 %0, %1, off"
                 :: "v"(l), "v"(g) : "memory");
}

__device__ __forceinline__ void wait_async_all() {
    asm volatile("s_wait_asynccnt 0" ::: "memory");
}

// ---------------- f32 -> f16 convert ----------------
__global__ void cvt_f32_f16(const float* __restrict__ s, _Float16* __restrict__ d, int n) {
    int i = blockIdx.x * blockDim.x + threadIdx.x;
    int stride = gridDim.x * blockDim.x;
    for (; i < n; i += stride) d[i] = (_Float16)s[i];
}

// ---------------- generic GEMM: C[M,N] = A[M,K] @ W[N,K]^T (+bias) ----------------
// blockDim = 256 (8 waves). Wave tile 32(M) x 64(N); block tile 256 x 64.
// B panel staged in LDS in K-slabs of 256 via async-to-LDS copies.
template <typename OutT, bool BIAS>
__global__ __launch_bounds__(256) void gemm_f16(const _Float16* __restrict__ A,
                                                const _Float16* __restrict__ W,
                                                OutT* __restrict__ C,
                                                const float* __restrict__ bias,
                                                int M, int N, int K, int ldc) {
    __shared__ _Float16 bsh[64 * BSTRIDE];

    const int tid  = threadIdx.x;
    const int wave = tid >> 5;
    const int lane = tid & 31;
    const int l15  = lane & 15;
    const int hi   = lane >> 4;

    const int m0 = blockIdx.x * 256 + wave * 32;
    const int n0 = blockIdx.y * 64;

    const _Float16* arow0 = A + (size_t)(m0 + l15) * K + hi * 8;
    const _Float16* arow1 = arow0 + (size_t)16 * K;
    const _Float16* wpanel = W + (size_t)n0 * K;

    v8f acc0[4], acc1[4];
#pragma unroll
    for (int t = 0; t < 4; ++t) { acc0[t] = (v8f)(0.0f); acc1[t] = (v8f)(0.0f); }

    for (int kk = 0; kk < K; kk += KC) {
        __syncthreads();   // protect previous slab reads
        // stage B slab: 64 rows x 256 halves = 2048 x 16B chunks
#pragma unroll
        for (int it = 0; it < 2048 / 256; ++it) {
            int c = tid + it * 256;
            int row = c >> 5;          // 0..63
            int col8 = c & 31;         // 8-half units within slab
            async_copy_b128(wpanel + (size_t)row * K + kk + col8 * 8,
                            bsh + row * BSTRIDE + col8 * 8);
        }
        wait_async_all();
        __syncthreads();

#pragma unroll
        for (int k0 = 0; k0 < KC; k0 += 32) {
            const int kg = kk + k0;
            v16h a0 = cat8(*(const v8h*)(arow0 + kg), *(const v8h*)(arow0 + kg + 16));
            v16h a1 = cat8(*(const v8h*)(arow1 + kg), *(const v8h*)(arow1 + kg + 16));
#pragma unroll
            for (int t = 0; t < 4; ++t) {
                const _Float16* bp = bsh + (t * 16 + l15) * BSTRIDE + hi * 8 + k0;
                v16h b = cat8(*(const v8h*)(bp), *(const v8h*)(bp + 16));
                acc0[t] = wmma_f16(a0, b, acc0[t]);
                acc1[t] = wmma_f16(a1, b, acc1[t]);
            }
        }
    }

    // C layout: VGPR r, lanes 0-15 -> (M=r, N=lane); lanes 16-31 -> (M=8+r, N=lane-16)
#pragma unroll
    for (int t = 0; t < 4; ++t) {
        int n = n0 + t * 16 + l15;
        float bv = 0.0f;
        if (BIAS) bv = bias[n];
#pragma unroll
        for (int r = 0; r < 8; ++r) {
            int m = m0 + hi * 8 + r;
            C[(size_t)m * ldc + n]        = (OutT)(acc0[t][r] + bv);
            C[(size_t)(m + 16) * ldc + n] = (OutT)(acc1[t][r] + bv);
        }
    }
}

// ---------------- attention pass A: per-row (max, sumexp) of S = scale * q @ k^T ----------------
__global__ __launch_bounds__(128) void attn_stats(const _Float16* __restrict__ q_h,   // [B*64, 768]
                                                  const _Float16* __restrict__ kv_h,  // [B*4096, 1536]
                                                  float* __restrict__ stats)          // [B*H*64][2]
{
    const int bh = blockIdx.x;
    const int b = bh / NUM_HEADS, h = bh % NUM_HEADS;
    const int wave = threadIdx.x >> 5;   // 0..3 -> d tile
    const int lane = threadIdx.x & 31;
    const int l15 = lane & 15, hi = lane >> 4;

    const _Float16* aptr = q_h + (size_t)(b * N1 + wave * 16 + l15) * BDIM + h * HD + hi * 16;
    v16h a0 = cat8(*(const v8h*)(aptr + 0),  *(const v8h*)(aptr + 8));
    v16h a1 = cat8(*(const v8h*)(aptr + 32), *(const v8h*)(aptr + 40));

    const _Float16* kb = kv_h + (size_t)b * N2 * 1536 + h * HD + hi * 16;

    float mrun[8], srun[8];
#pragma unroll
    for (int r = 0; r < 8; ++r) { mrun[r] = -3.0e38f; srun[r] = 0.0f; }

    for (int n2c = 0; n2c < N2; n2c += 16) {
        const _Float16* kp = kb + (size_t)(n2c + l15) * 1536;
        v16h b0 = *(const v16h*)(kp);
        v16h b1 = *(const v16h*)(kp + 32);
        v8f acc = (v8f)(0.0f);
        acc = wmma_f16(a0, b0, acc);
        acc = wmma_f16(a1, b1, acc);
#pragma unroll
        for (int r = 0; r < 8; ++r) {
            float s = acc[r] * SCALE;
            float nm = fmaxf(mrun[r], s);
            srun[r] = srun[r] * __expf(mrun[r] - nm) + __expf(s - nm);
            mrun[r] = nm;
        }
    }

#pragma unroll
    for (int mask = 1; mask < 16; mask <<= 1) {
#pragma unroll
        for (int r = 0; r < 8; ++r) {
            float om = __shfl_xor(mrun[r], mask);
            float os = __shfl_xor(srun[r], mask);
            float nm = fmaxf(mrun[r], om);
            srun[r] = srun[r] * __expf(mrun[r] - nm) + os * __expf(om - nm);
            mrun[r] = nm;
        }
    }

    if (l15 == 0) {
#pragma unroll
        for (int r = 0; r < 8; ++r) {
            int d = wave * 16 + hi * 8 + r;
            float* sp = stats + ((size_t)bh * N1 + d) * 2;
            sp[0] = mrun[r];
            sp[1] = srun[r];
        }
    }
}

// ---------------- attention pass B ----------------
__global__ __launch_bounds__(256) void attn_apply(const _Float16* __restrict__ q_h,
                                                  const _Float16* __restrict__ kv_h,
                                                  const float* __restrict__ stats,
                                                  _Float16* __restrict__ z_h)  // [B*4096, 768]
{
    const int bh = blockIdx.y;
    const int b = bh / NUM_HEADS, h = bh % NUM_HEADS;
    const int wave = threadIdx.x >> 5;
    const int d_tile = wave & 3;
    const int nsub = wave >> 2;
    const int lane = threadIdx.x & 31;
    const int l15 = lane & 15, hi = lane >> 4;
    const int n2base = blockIdx.x * 32 + nsub * 16;

    const _Float16* aptr = q_h + (size_t)(b * N1 + d_tile * 16 + l15) * BDIM + h * HD + hi * 16;
    v16h a0 = cat8(*(const v8h*)(aptr + 0),  *(const v8h*)(aptr + 8));
    v16h a1 = cat8(*(const v8h*)(aptr + 32), *(const v8h*)(aptr + 40));

    const _Float16* kp = kv_h + (size_t)(b * N2 + n2base + l15) * 1536 + h * HD + hi * 16;
    v16h b0 = *(const v16h*)(kp);
    v16h b1 = *(const v16h*)(kp + 32);

    v8f acc = (v8f)(0.0f);
    acc = wmma_f16(a0, b0, acc);
    acc = wmma_f16(a1, b1, acc);

    const int n2 = n2base + l15;
    const _Float16* vp = kv_h + (size_t)(b * N2 + n2) * 1536 + BDIM + h * HD + d_tile * 16 + hi * 8;
    v8h vv = *(const v8h*)vp;

    v8h zout;
#pragma unroll
    for (int r = 0; r < 8; ++r) {
        int d = d_tile * 16 + hi * 8 + r;
        const float* sp = stats + ((size_t)bh * N1 + d) * 2;
        float p = __expf(acc[r] * SCALE - sp[0]) / sp[1];
        zout[r] = (_Float16)(p * (float)vv[r]);
    }
    *(v8h*)(z_h + (size_t)(b * N2 + n2) * BDIM + h * HD + d_tile * 16 + hi * 8) = zout;
}

extern "C" void kernel_launch(void* const* d_in, const int* in_sizes, int n_in,
                              void* d_out, int out_size, void* d_ws, size_t ws_size,
                              hipStream_t stream) {
    (void)in_sizes; (void)n_in; (void)out_size; (void)ws_size;
    const float* x      = (const float*)d_in[0];
    const float* y      = (const float*)d_in[1];
    const float* W_qkv  = (const float*)d_in[2];
    const float* W_proj = (const float*)d_in[3];
    const float* b_proj = (const float*)d_in[4];
    float* out = (float*)d_out;

    char* p = (char*)d_ws;
    auto take = [&](size_t bytes) -> void* {
        void* r = (void*)p;
        p += (bytes + 255) & ~(size_t)255;
        return r;
    };
    _Float16* wqkv_h  = (_Float16*)take((size_t)3 * BDIM * BDIM * 2);
    _Float16* wproj_h = (_Float16*)take((size_t)BDIM * BDIM * 2);
    _Float16* x_h     = (_Float16*)take((size_t)NB * N1 * BDIM * 2);
    _Float16* y_h     = (_Float16*)take((size_t)NB * N2 * BDIM * 2);
    _Float16* q_h     = (_Float16*)take((size_t)NB * N1 * BDIM * 2);
    _Float16* kv_h    = (_Float16*)take((size_t)NB * N2 * 2 * BDIM * 2);
    _Float16* z_h     = (_Float16*)take((size_t)NB * N2 * BDIM * 2);
    float*    stats   = (float*)take((size_t)NB * NUM_HEADS * N1 * 2 * 4);

    auto cvt_grid = [](int n) { int g = (n + 255) / 256; return g > 4096 ? 4096 : g; };
    int n_wqkv = 3 * BDIM * BDIM, n_wp = BDIM * BDIM;
    int n_x = NB * N1 * BDIM, n_y = NB * N2 * BDIM;
    cvt_f32_f16<<<cvt_grid(n_wqkv), 256, 0, stream>>>(W_qkv, wqkv_h, n_wqkv);
    cvt_f32_f16<<<cvt_grid(n_wp),   256, 0, stream>>>(W_proj, wproj_h, n_wp);
    cvt_f32_f16<<<cvt_grid(n_x),    256, 0, stream>>>(x, x_h, n_x);
    cvt_f32_f16<<<cvt_grid(n_y),    256, 0, stream>>>(y, y_h, n_y);

    // q = x @ W_q^T               [512,768]
    gemm_f16<_Float16, false><<<dim3(512 / 256, BDIM / 64), 256, 0, stream>>>(
        x_h, wqkv_h, q_h, nullptr, 512, BDIM, BDIM, BDIM);
    // kv = y @ W_kv^T             [32768,1536] (cols 0..767 = k, 768..1535 = v)
    gemm_f16<_Float16, false><<<dim3((NB * N2) / 256, (2 * BDIM) / 64), 256, 0, stream>>>(
        y_h, wqkv_h + (size_t)BDIM * BDIM, kv_h, nullptr, NB * N2, 2 * BDIM, BDIM, 2 * BDIM);

    attn_stats<<<NB * NUM_HEADS, 128, 0, stream>>>(q_h, kv_h, stats);
    attn_apply<<<dim3(N2 / 32, NB * NUM_HEADS), 256, 0, stream>>>(q_h, kv_h, stats, z_h);

    // out = Z @ W_proj^T + b      [32768,768] f32
    gemm_f16<float, true><<<dim3((NB * N2) / 256, BDIM / 64), 256, 0, stream>>>(
        z_h, wproj_h, out, b_proj, NB * N2, BDIM, BDIM, BDIM);
}
